// AttentionHead_6365141532793
// MI455X (gfx1250) — matrix operand order repaired and checked by reading, not compiled, not run on hardware
//
#include <hip/hip_runtime.h>

// ---------------------------------------------------------------------------
// AttentionHead for MI455X (gfx1250, wave32, WMMA + async global->LDS).
//   q = query*Wq^T + b ; k = key*Wq^T + b ; v = value*Wq^T + b   (f16 outputs)
//   out = softmax(q k^T / 16) v                                   (f32 accum)
// ---------------------------------------------------------------------------

typedef __attribute__((ext_vector_type(16))) _Float16 v16h;
typedef __attribute__((ext_vector_type(8)))  _Float16 v8h;
typedef __attribute__((ext_vector_type(8)))  float    v8f;
typedef __attribute__((ext_vector_type(4)))  float    v4f;

#define B_ 4
#define S_ 4096
#define D_ 256

template <bool RA>
static __device__ __forceinline__ v8f wmma_f16(v16h a, v16h b, v8f c) {
    return __builtin_amdgcn_wmma_f32_16x16x32_f16(
        /*neg_a=*/false, a, /*neg_b=*/false, b,
        /*c_mod=*/(short)0, c, /*reuse_a=*/RA, /*reuse_b=*/false);
}

// Build v16h A/B fragment from two contiguous 8-half chunks.
static __device__ __forceinline__ v16h frag_from_halves(const _Float16* lo_p,
                                                        const _Float16* hi_p) {
    v8h lo = *(const v8h*)lo_p;
    v8h hi = *(const v8h*)hi_p;
    v16h f;
#pragma unroll
    for (int i = 0; i < 8; ++i) { f[i] = lo[i]; f[8 + i] = hi[i]; }
    return f;
}

// Async global -> LDS copy, 16 bytes per lane (CDNA5, tracked by ASYNCcnt).
// Low 32 bits of a generic pointer to __shared__ are the LDS byte offset.
static __device__ __forceinline__ void async_copy_b128(const _Float16* lds_dst,
                                                       const _Float16* gsrc) {
    uint32_t lds_off = (uint32_t)(size_t)(const void*)lds_dst;
    unsigned long long ga = (unsigned long long)(size_t)(const void*)gsrc;
    asm volatile("global_load_async_to_lds_b128 %0, %1, off"
                 :: "v"(lds_off), "v"(ga) : "memory");
}

// ---------------------------------------------------------------------------
// Projection: Y[m][n] = sum_d X[m][d] * W[n][d] + bias[n]  (f32 in, f16 out)
// One wave computes a 16x16 output tile. transpose=1 writes Y as [B, D, S].
// ---------------------------------------------------------------------------
__global__ __launch_bounds__(32) void proj_kernel(
    const float* __restrict__ X, const float* __restrict__ W,
    const float* __restrict__ bias, _Float16* __restrict__ Y, int transpose)
{
    const int lane = threadIdx.x & 31;
    const int ln = lane & 15;
    const int hi = lane >> 4;
    const int m0 = blockIdx.x * 16;          // global row in [0, B*S)
    const int n0 = blockIdx.y * 16;          // output column
    const int bb = m0 / S_;
    const int s0 = m0 % S_;

    const float* arow = X + (size_t)(m0 + ln) * D_;   // A: row = m0+ln
    const float* brow = W + (size_t)(n0 + ln) * D_;   // B: col = n0+ln (W row)

    v8f c;
#pragma unroll
    for (int r = 0; r < 8; ++r) c[r] = 0.0f;

#pragma unroll
    for (int ch = 0; ch < 8; ++ch) {
        // A fragment: 16 consecutive K starting at ch*32 + 16*hi
        const float* ap = arow + ch * 32 + 16 * hi;
        v4f a0 = *(const v4f*)(ap);
        v4f a1 = *(const v4f*)(ap + 4);
        v4f a2 = *(const v4f*)(ap + 8);
        v4f a3 = *(const v4f*)(ap + 12);
        v16h af;
#pragma unroll
        for (int i = 0; i < 4; ++i) {
            af[i]      = (_Float16)a0[i];
            af[4 + i]  = (_Float16)a1[i];
            af[8 + i]  = (_Float16)a2[i];
            af[12 + i] = (_Float16)a3[i];
        }
        // B fragment: 16 consecutive K starting at ch*32 + 16*hi
        const float* bp = brow + ch * 32 + 16 * hi;
        v16h bf;
#pragma unroll
        for (int i = 0; i < 16; i += 4) {
            v4f t = *(const v4f*)(bp + i);
            bf[i]     = (_Float16)t[0];
            bf[i + 1] = (_Float16)t[1];
            bf[i + 2] = (_Float16)t[2];
            bf[i + 3] = (_Float16)t[3];
        }
        c = wmma_f16<false>(af, bf, c);
    }

    const float bv = bias[n0 + ln];
#pragma unroll
    for (int r = 0; r < 8; ++r) {
        _Float16 h = (_Float16)(c[r] + bv);
        const int rr  = r + 8 * hi;          // row within tile
        const int col = n0 + ln;
        if (!transpose)
            Y[((size_t)bb * S_ + s0 + rr) * D_ + col] = h;
        else
            Y[((size_t)bb * D_ + col) * S_ + s0 + rr] = h;   // V^T [B,D,S]
    }
}

// ---------------------------------------------------------------------------
// Flash attention. Block = 4 waves = 64 query rows (16 per wave).
// Per iteration (32 keys): K tile (32x256 f16) and V^T tile (256x32 f16) are
// staged into double-buffered LDS with async global->LDS copies (issued one
// iteration ahead, drained with s_wait_asynccnt), then consumed as WMMA
// fragments via ds_load_b128. Online softmax runs in the log2 domain.
// ---------------------------------------------------------------------------
__global__ __launch_bounds__(128, 1) void attn_kernel(
    const _Float16* __restrict__ Q, const _Float16* __restrict__ K,
    const _Float16* __restrict__ VT, float* __restrict__ O)
{
    __shared__ alignas(16) _Float16 kbufs[2][32 * 256];  // 2 x 16 KB
    __shared__ alignas(16) _Float16 vbufs[2][256 * 32];  // 2 x 16 KB
    __shared__ alignas(16) _Float16 pbuf[4][16 * 32];    // per-wave P tile

    const int tid  = threadIdx.x;          // 0..127
    const int lane = tid & 31;
    const int w    = tid >> 5;
    const int ln = lane & 15;
    const int hi = lane >> 4;
    const int b  = blockIdx.y;
    const int q0 = (blockIdx.x * 4 + w) * 16;

    // Issue async copies for one 32-key tile into LDS buffer `buf`.
    auto issue_tile = [&](int kb, int buf) {
        // K tile: 32 rows x 256 halfs = 1024 x 16B chunks, 8 per thread.
#pragma unroll
        for (int j = 0; j < 8; ++j) {
            const int c   = tid + j * 128;
            const int row = c >> 5;
            const int dq  = (c & 31) * 8;
            async_copy_b128(&kbufs[buf][row * 256 + dq],
                            K + ((size_t)b * S_ + kb + row) * D_ + dq);
        }
        // V^T tile: 256 rows x 32 halfs = 1024 x 16B chunks, 8 per thread.
#pragma unroll
        for (int j = 0; j < 8; ++j) {
            const int c  = tid + j * 128;
            const int d  = c >> 2;
            const int sq = (c & 3) * 8;
            async_copy_b128(&vbufs[buf][d * 32 + sq],
                            VT + ((size_t)b * D_ + d) * S_ + kb + sq);
        }
    };

    // Q fragments for all 8 K-chunks (row = q0+ln, held for the whole kernel)
    v16h qf[8];
    {
        const _Float16* qrow = Q + ((size_t)b * S_ + q0 + ln) * D_;
#pragma unroll
        for (int ch = 0; ch < 8; ++ch) {
            const _Float16* p = qrow + ch * 32 + 16 * hi;
            qf[ch] = frag_from_halves(p, p + 8);
        }
    }

    v8f acc[16];
#pragma unroll
    for (int t = 0; t < 16; ++t)
#pragma unroll
        for (int r = 0; r < 8; ++r) acc[t][r] = 0.0f;

    float m[8], l[8];
#pragma unroll
    for (int r = 0; r < 8; ++r) { m[r] = -3.0e38f; l[r] = 0.0f; }

    // softmax in log2 domain: exp(s/16) == exp2(s * (log2e/16))
    const float cscale = 0.0625f * 1.44269504088896f;

    issue_tile(0, 0);                                   // prologue

    for (int it = 0; it < S_ / 32; ++it) {
        const int kb = it * 32;
        if (kb + 32 < S_) {
            issue_tile(kb + 32, (it + 1) & 1);          // prefetch next tile
            asm volatile("s_wait_asynccnt 0x10" ::: "memory");  // tile it done
        } else {
            asm volatile("s_wait_asynccnt 0x0" ::: "memory");
        }
        __syncthreads();                                // tile it visible to all

        const _Float16* kbuf = kbufs[it & 1];
        const _Float16* vbuf = vbufs[it & 1];

        // ---- scores: two 16x16 tiles (keys kb..kb+15, kb+16..kb+31) ----
        v8f s0v, s1v;
#pragma unroll
        for (int r = 0; r < 8; ++r) { s0v[r] = 0.0f; s1v[r] = 0.0f; }

#pragma unroll
        for (int ch = 0; ch < 8; ++ch) {
            const _Float16* kp0 = kbuf + ln * D_ + ch * 32 + 16 * hi;
            v16h bf0 = frag_from_halves(kp0, kp0 + 8);
            s0v = wmma_f16<false>(qf[ch], bf0, s0v);
            const _Float16* kp1 = kp0 + 16 * D_;
            v16h bf1 = frag_from_halves(kp1, kp1 + 8);
            s1v = wmma_f16<false>(qf[ch], bf1, s1v);
        }

        // ---- online softmax (rows r and r+8 per lane group) ----
        float alpha[8];
#pragma unroll
        for (int r = 0; r < 8; ++r) {
            float a  = s0v[r] * cscale;
            float c2 = s1v[r] * cscale;
            float mx = fmaxf(a, c2);
#pragma unroll
            for (int msk = 1; msk < 16; msk <<= 1)
                mx = fmaxf(mx, __shfl_xor(mx, msk, 32));
            const float mn = fmaxf(m[r], mx);
            const float al = __builtin_amdgcn_exp2f(m[r] - mn);
            const float p0 = __builtin_amdgcn_exp2f(a - mn);
            const float p1 = __builtin_amdgcn_exp2f(c2 - mn);
            float rs = p0 + p1;
#pragma unroll
            for (int msk = 1; msk < 16; msk <<= 1)
                rs += __shfl_xor(rs, msk, 32);
            l[r] = l[r] * al + rs;
            m[r] = mn;
            alpha[r] = al;
            const int rr = r + 8 * hi;
            pbuf[w][rr * 32 + ln]      = (_Float16)p0;
            pbuf[w][rr * 32 + 16 + ln] = (_Float16)p1;
        }

        // rescale accumulator
#pragma unroll
        for (int t = 0; t < 16; ++t)
#pragma unroll
            for (int r = 0; r < 8; ++r) acc[t][r] *= alpha[r];

        // ---- P (16x32 f16) as A-fragment from LDS ----
        const _Float16* pp = &pbuf[w][ln * 32 + 16 * hi];
        v16h pf = frag_from_halves(pp, pp + 8);

        // ---- acc += P * V : B-fragments from staged V^T tile ----
        {
            const _Float16* vp0 = vbuf + ln * 32 + 16 * hi;
            v16h vf0 = frag_from_halves(vp0, vp0 + 8);
            acc[0] = wmma_f16<false>(pf, vf0, acc[0]);
        }
#pragma unroll
        for (int t = 1; t < 16; ++t) {
            const _Float16* vp = vbuf + (t * 16 + ln) * 32 + 16 * hi;
            v16h vf = frag_from_halves(vp, vp + 8);
            acc[t] = wmma_f16<true>(pf, vf, acc[t]);   // reuse A (same pf)
        }

        __syncthreads();    // all waves done with buffer (it&1) before refill
    }

    // ---- epilogue: normalize and store fp32 ----
#pragma unroll
    for (int t = 0; t < 16; ++t)
#pragma unroll
        for (int r = 0; r < 8; ++r) {
            const int rr = r + 8 * hi;
            O[((size_t)b * S_ + q0 + rr) * D_ + t * 16 + ln] =
                acc[t][r] / l[r];
        }
}

// ---------------------------------------------------------------------------
extern "C" void kernel_launch(void* const* d_in, const int* in_sizes, int n_in,
                              void* d_out, int out_size, void* d_ws, size_t ws_size,
                              hipStream_t stream) {
    (void)in_sizes; (void)n_in; (void)out_size; (void)ws_size;
    const float* query = (const float*)d_in[0];
    const float* key   = (const float*)d_in[1];
    const float* value = (const float*)d_in[2];
    const float* Wq    = (const float*)d_in[3];
    const float* bq    = (const float*)d_in[4];
    float* out = (float*)d_out;

    const size_t n_elem = (size_t)B_ * S_ * D_;
    _Float16* qh = (_Float16*)d_ws;
    _Float16* kh = qh + n_elem;
    _Float16* vT = kh + n_elem;   // stored [B, D, S]

    dim3 pg(B_ * S_ / 16, D_ / 16);
    proj_kernel<<<pg, 32, 0, stream>>>(query, Wq, bq, qh, 0);
    proj_kernel<<<pg, 32, 0, stream>>>(key,   Wq, bq, kh, 0);
    proj_kernel<<<pg, 32, 0, stream>>>(value, Wq, bq, vT, 1);

    dim3 ag(S_ / 64, B_);
    attn_kernel<<<ag, 128, 0, stream>>>(qh, kh, vT, out);
}
